// Attention_26018911879416
// MI455X (gfx1250) — matrix operand order repaired and checked by reading, not compiled, not run on hardware
//
#include <hip/hip_runtime.h>

typedef __attribute__((ext_vector_type(16))) _Float16 v16h;
typedef __attribute__((ext_vector_type(8)))  _Float16 v8h;
typedef __attribute__((ext_vector_type(8)))  float    v8f;
typedef __attribute__((ext_vector_type(4)))  unsigned u32x4;
typedef __attribute__((ext_vector_type(8)))  unsigned u32x8;

#define D_MODEL 1024
#define N_HEADS 16
#define D_HEAD  64
#define BATCH   4
#define SEQ     2048
#define BT      (BATCH*SEQ)   /* 8192 */

// ---------------------------------------------------------------------------
// gfx1250 LDS address: the shared aperture keeps the byte offset in
// addr[31:0], so truncating the flat pointer yields the LDS address.
// ---------------------------------------------------------------------------
__device__ __forceinline__ unsigned lds_addr32(const void* p) {
    return (unsigned)(unsigned long long)p;
}

// ---------------------------------------------------------------------------
// Async global->LDS (ASYNCcnt path, CDNA5 ISA)
// ---------------------------------------------------------------------------
__device__ __forceinline__ void async_ld16(void* lds_dst, const void* gsrc) {
    asm volatile("global_load_async_to_lds_b128 %0, %1, off"
                 :: "v"(lds_addr32(lds_dst)), "v"(gsrc)
                 : "memory");
}
__device__ __forceinline__ void wait_async0() {
    asm volatile("s_wait_asynccnt 0x0" ::: "memory");
}

// ---------------------------------------------------------------------------
// Tensor Data Mover: 2D tile load, data_size = 8B units (CDNA5 ISA-8).
// D# group0: [1:0]=count=1, [63:32]=lds_addr, [120:64]=global_addr,
//            [127:126]=type=2.  D# group1: [17:16]=data_size(3 -> 8B),
//            [79:48]=tensor_dim0, [111:80]=tensor_dim1, [127:112]=tile_dim0,
//            [143:128]=tile_dim1, [207:160]=tensor_dim0_stride.
// Tile rows land contiguously in LDS (row-major), tensor dims == tile dims.
// All inputs are block-uniform -> descriptors stay in SGPRs.
// ---------------------------------------------------------------------------
__device__ __forceinline__ void tdm_load_2d(unsigned lds, const void* g,
                                            unsigned tileX, unsigned tileY,
                                            unsigned strideX) {
    unsigned long long ga = (unsigned long long)g;
    u32x4 g0;
    g0[0] = 1u;                                   // count=1, user mode
    g0[1] = lds;                                  // lds_addr (bytes)
    g0[2] = (unsigned)ga;                         // global_addr[31:0]
    g0[3] = (unsigned)(ga >> 32) | 0x80000000u;   // global_addr[56:32] | type=2
    u32x8 g1;
    g1[0] = (3u << 16);                           // data_size = 8B, no mask
    g1[1] = (tileX & 0xFFFFu) << 16;              // tensor_dim0[15:0]
    g1[2] = ((tileX >> 16) & 0xFFFFu)             // tensor_dim0[31:16]
          | ((tileY & 0xFFFFu) << 16);            // tensor_dim1[15:0]
    g1[3] = ((tileY >> 16) & 0xFFFFu)             // tensor_dim1[31:16]
          | ((tileX & 0xFFFFu) << 16);            // tile_dim0
    g1[4] = (tileY & 0xFFFFu);                    // tile_dim1, tile_dim2=0
    g1[5] = strideX;                              // tensor_dim0_stride[31:0]
    g1[6] = 0u;
    g1[7] = 0u;
    asm volatile("tensor_load_to_lds %0, %1" :: "s"(g0), "s"(g1) : "memory");
}

// ---------------------------------------------------------------------------
// f32 -> f16 conversion (straight)
// ---------------------------------------------------------------------------
__global__ void cvt_f32_to_f16(const float* __restrict__ in,
                               _Float16* __restrict__ out, int n) {
    int i = blockIdx.x * blockDim.x + threadIdx.x;
    if (i < n) out[i] = (_Float16)in[i];
}

// ---------------------------------------------------------------------------
// f32 -> f16 with transpose: in[R][C] -> out[C][R]  (weights, done once)
// ---------------------------------------------------------------------------
__global__ void cvt_transpose_f32_to_f16(const float* __restrict__ in,
                                         _Float16* __restrict__ out,
                                         int R, int C) {
    int i = blockIdx.x * blockDim.x + threadIdx.x;
    if (i >= R * C) return;
    int r = i / C, c = i % C;
    out[(size_t)c * R + r] = (_Float16)in[i];
}

// ---------------------------------------------------------------------------
// Tiled WMMA GEMM: C[M,N] = A[M,K] * Bt[N,K]^T (both f16 row-major as stored).
// Block tile 128x128, BK=32. 256 threads = 8 waves (4x2), each wave 32x64.
// A/B tiles staged by the Tensor Data Mover, double buffered.
// MODE: 0 = f32 out, 1 = f16 out, 2 = f16 scatter into [b][h][d][t] (for V)
// ---------------------------------------------------------------------------
template <int MODE>
__global__ __launch_bounds__(256)
void gemm_f16(const _Float16* __restrict__ A, const _Float16* __restrict__ Bt,
              void* __restrict__ Cout, int M, int N, int K) {
    __shared__ __align__(16) _Float16 As[2][128][32];   // [buf][m][k]
    __shared__ __align__(16) _Float16 Bs[2][128][32];   // [buf][n][k]

    const int tid  = threadIdx.x;
    const int wave = tid >> 5, lane = tid & 31;
    const int wm = wave >> 1, wn = wave & 1;
    const int lrow = lane & 15;
    const int hi   = lane >> 4;                          // 0 or 1
    const int blockM = blockIdx.y * 128;
    const int blockN = blockIdx.x * 128;
    const bool w0 = (__builtin_amdgcn_readfirstlane(wave) == 0);

    auto stage = [&](int buf, int k0) {
        if (w0) {  // one TDM op per tile replaces ~512 per-lane loads
            tdm_load_2d(lds_addr32(&As[buf][0][0]),
                        A + (size_t)blockM * K + k0, 8, 128, (unsigned)(K / 4));
            tdm_load_2d(lds_addr32(&Bs[buf][0][0]),
                        Bt + (size_t)blockN * K + k0, 8, 128, (unsigned)(K / 4));
        }
    };

    v8f acc[2][4] = {};

    stage(0, 0);
    const int nsteps = K / 32;
    for (int ki = 0; ki < nsteps; ki++) {
        const int cur = ki & 1;
        if (w0) __builtin_amdgcn_s_wait_tensorcnt(0);
        __syncthreads();
        if (ki + 1 < nsteps) stage(1 - cur, (ki + 1) * 32);

        // ---- fragments (16-bit WMMA lane layouts) ----
        v16h afrag[2], bfrag[4];
        #pragma unroll
        for (int mt = 0; mt < 2; mt++) {
            int r  = wm * 32 + mt * 16 + lrow;
            int kb = hi * 8;                             // A: K0..7/16..23 | K8..15/24..31
            union { v16h v; v8h h8[2]; } u;
            u.h8[0] = *(const v8h*)(&As[cur][r][kb]);
            u.h8[1] = *(const v8h*)(&As[cur][r][kb + 16]);
            afrag[mt] = u.v;
        }
        #pragma unroll
        for (int nt = 0; nt < 4; nt++) {
            int n  = wn * 64 + nt * 16 + lrow;
            int kb = hi * 8;                             // B: K0..7/16..23 | K8..15/24..31
            union { v16h v; v8h h8[2]; } u;
            u.h8[0] = *(const v8h*)(&Bs[cur][n][kb]);
            u.h8[1] = *(const v8h*)(&Bs[cur][n][kb + 16]);
            bfrag[nt] = u.v;
        }
        #pragma unroll
        for (int mt = 0; mt < 2; mt++)
            #pragma unroll
            for (int nt = 0; nt < 4; nt++)
                acc[mt][nt] = __builtin_amdgcn_wmma_f32_16x16x32_f16(
                    false, afrag[mt], false, bfrag[nt],
                    (short)0, acc[mt][nt], false, false);
        __syncthreads();
    }

    // ---- epilogue: C layout (VGPR i, lane) -> row i(+8), col lane%16 ----
    #pragma unroll
    for (int mt = 0; mt < 2; mt++)
        #pragma unroll
        for (int nt = 0; nt < 4; nt++)
            #pragma unroll
            for (int i = 0; i < 8; i++) {
                size_t r = blockM + wm * 32 + mt * 16 + i + hi * 8;
                size_t c = blockN + wn * 64 + nt * 16 + lrow;
                float v = acc[mt][nt][i];
                if (MODE == 0) {
                    ((float*)Cout)[r * N + c] = v;
                } else if (MODE == 1) {
                    ((_Float16*)Cout)[r * N + c] = (_Float16)v;
                } else {
                    int bb = (int)(r / SEQ), t = (int)(r % SEQ);
                    int hh = (int)(c / D_HEAD), d = (int)(c % D_HEAD);
                    ((_Float16*)Cout)[(((size_t)(bb * N_HEADS + hh)) * D_HEAD + d)
                                      * SEQ + t] = (_Float16)v;
                }
            }
}

// ---------------------------------------------------------------------------
// Flash attention (causal). One block = 64 query rows of one (b,h);
// 4 waves x 16 rows, d_head = 64. K tile staged by TDM, V tile by async
// copies; both double buffered in LDS and shared by all 4 waves.
// ---------------------------------------------------------------------------
__global__ __launch_bounds__(128)
void flash_attn(const _Float16* __restrict__ Q, const _Float16* __restrict__ Kmat,
                const _Float16* __restrict__ Vt, _Float16* __restrict__ Z) {
    __shared__ __align__(16) _Float16 Kt[2][32][64];     // [buf][key][d]
    __shared__ __align__(16) _Float16 Vs[2][64][32];     // [buf][d][key]
    __shared__ __align__(16) _Float16 Plds[4][16][32];   // per-wave P slab

    const int nqb  = SEQ / 64;                           // 32
    const int qblk = blockIdx.x % nqb;
    const int h    = (blockIdx.x / nqb) % N_HEADS;
    const int b    = blockIdx.x / (nqb * N_HEADS);
    const int wave = threadIdx.x >> 5, lane = threadIdx.x & 31;
    const int lrow = lane & 15, hi = lane >> 4;
    const int q0   = qblk * 64 + wave * 16;
    const bool w0  = (__builtin_amdgcn_readfirstlane(wave) == 0);

    const _Float16* Kbase = Kmat + (size_t)(b * SEQ) * D_MODEL + h * D_HEAD;
    const _Float16* Vbase = Vt + ((size_t)(b * N_HEADS + h)) * D_HEAD * SEQ;

    auto stage = [&](int buf, int kb0) {
        if (w0) {   // K tile 32x64 f16: one TDM 2D descriptor
            tdm_load_2d(lds_addr32(&Kt[buf][0][0]),
                        Kbase + (size_t)kb0 * D_MODEL,
                        16, 32, (unsigned)(D_MODEL / 4));
        }
        #pragma unroll
        for (int s = 0; s < 2; s++) {                    // V tile: async path
            int seg = threadIdx.x + s * 128;             // 0..255
            int d  = seg >> 2;                           // 0..63
            int ks = (seg & 3) * 8;                      // 0,8,16,24
            async_ld16(&Vs[buf][d][ks], Vbase + (size_t)d * SEQ + kb0 + ks);
        }
    };

    // Q fragments (A layout): lane = q row, K-dim along d_head, 2 halves
    v16h qf[2];
    {
        const _Float16* qrow =
            Q + ((size_t)(b * SEQ) + q0 + lrow) * D_MODEL + h * D_HEAD;
        #pragma unroll
        for (int kf = 0; kf < 2; kf++) {
            int kb = kf * 32 + hi * 8;
            union { v16h v; v8h h8[2]; } u;
            u.h8[0] = *(const v8h*)(qrow + kb);
            u.h8[1] = *(const v8h*)(qrow + kb + 16);
            qf[kf] = u.v;
        }
    }

    float m[8], l[8];
    v8f zacc[4] = {};
    #pragma unroll
    for (int i = 0; i < 8; i++) { m[i] = -3.0e38f; l[i] = 0.f; }

    // base-2 softmax: fold 1/sqrt(64) * log2(e) into one scale
    const float sc2 = 0.125f * 1.44269504088896340736f;
    const int kend_s = __builtin_amdgcn_readfirstlane(q0 + 16); // scalar bound
    const int nsteps = (qblk * 64 + 64) / 32;            // block-uniform

    stage(0, 0);
    for (int ki = 0; ki < nsteps; ki++) {
        const int kb0 = ki * 32;
        const int cur = ki & 1;
        wait_async0();
        if (w0) __builtin_amdgcn_s_wait_tensorcnt(0);
        __syncthreads();
        if (ki + 1 < nsteps) stage(1 - cur, kb0 + 32);

        if (kb0 < kend_s) {   // scalar branch: EXEC untouched for WMMA
            // ---- S = Q K^T ----
            v16h kfrag[2][2];
            #pragma unroll
            for (int t = 0; t < 2; t++) {
                const _Float16* krow = &Kt[cur][t * 16 + lrow][0];
                #pragma unroll
                for (int kf = 0; kf < 2; kf++) {
                    int db = kf * 32 + hi * 8;
                    union { v16h v; v8h h8[2]; } u;
                    u.h8[0] = *(const v8h*)(krow + db);
                    u.h8[1] = *(const v8h*)(krow + db + 16);
                    kfrag[t][kf] = u.v;
                }
            }
            v8f s[2] = {};
            #pragma unroll
            for (int t = 0; t < 2; t++)
                #pragma unroll
                for (int kf = 0; kf < 2; kf++)
                    s[t] = __builtin_amdgcn_wmma_f32_16x16x32_f16(
                        false, qf[kf], false, kfrag[t][kf],
                        (short)0, s[t], false, false);

            // ---- V fragments early (overlap with softmax VALU) ----
            v16h vfrag[4];
            #pragma unroll
            for (int nt = 0; nt < 4; nt++) {
                const _Float16* vrow = &Vs[cur][nt * 16 + lrow][hi * 8];
                union { v16h v; v8h h8[2]; } u;
                u.h8[0] = *(const v8h*)(vrow);
                u.h8[1] = *(const v8h*)(vrow + 16);
                vfrag[nt] = u.v;
            }

            // ---- causal mask + online softmax (base 2) ----
            float p[2][8];
            #pragma unroll
            for (int i = 0; i < 8; i++) {
                int row = q0 + i + hi * 8;
                float rowmax = -3.0e38f;
                #pragma unroll
                for (int t = 0; t < 2; t++) {
                    int col = kb0 + t * 16 + lrow;
                    float sv = (col <= row) ? s[t][i] * sc2 : -1.0e30f;
                    p[t][i] = sv;
                    rowmax = fmaxf(rowmax, sv);
                }
                #pragma unroll
                for (int off = 1; off < 16; off <<= 1)
                    rowmax = fmaxf(rowmax, __shfl_xor(rowmax, off, 16));
                float mnew = fmaxf(m[i], rowmax);
                float corr = exp2f(m[i] - mnew);
                float rs = 0.f;
                #pragma unroll
                for (int t = 0; t < 2; t++) {
                    p[t][i] = exp2f(p[t][i] - mnew);
                    rs += p[t][i];
                }
                #pragma unroll
                for (int off = 1; off < 16; off <<= 1)
                    rs += __shfl_xor(rs, off, 16);
                l[i] = l[i] * corr + rs;
                m[i] = mnew;
                #pragma unroll
                for (int nt = 0; nt < 4; nt++) zacc[nt][i] *= corr;
            }

            // ---- C-layout P -> row-major LDS (wave-local) -> A fragment ----
            #pragma unroll
            for (int t = 0; t < 2; t++)
                #pragma unroll
                for (int i = 0; i < 8; i++)
                    Plds[wave][i + hi * 8][t * 16 + lrow] = (_Float16)p[t][i];
            asm volatile("s_wait_dscnt 0" ::: "memory");  // wave32 lockstep sync
            v16h pf;
            {
                int kb = hi * 8;
                union { v16h v; v8h h8[2]; } u;
                u.h8[0] = *(const v8h*)(&Plds[wave][lrow][kb]);
                u.h8[1] = *(const v8h*)(&Plds[wave][lrow][kb + 16]);
                pf = u.v;
            }

            // ---- Z += P * V ----
            #pragma unroll
            for (int nt = 0; nt < 4; nt++)
                zacc[nt] = __builtin_amdgcn_wmma_f32_16x16x32_f16(
                    false, pf, false, vfrag[nt], (short)0, zacc[nt], false, false);
        }
        __syncthreads();
    }

    // ---- normalize and store Z (f16, heads packed) ----
    #pragma unroll
    for (int nt = 0; nt < 4; nt++)
        #pragma unroll
        for (int i = 0; i < 8; i++) {
            size_t row = q0 + i + hi * 8;
            float zv = zacc[nt][i] / l[i];
            Z[((size_t)(b * SEQ) + row) * D_MODEL + h * D_HEAD + nt * 16 + lrow] =
                (_Float16)zv;
        }
}

// ---------------------------------------------------------------------------
// Launch
// ---------------------------------------------------------------------------
extern "C" void kernel_launch(void* const* d_in, const int* in_sizes, int n_in,
                              void* d_out, int out_size, void* d_ws, size_t ws_size,
                              hipStream_t stream) {
    const float* x  = (const float*)d_in[0];
    const float* Wq = (const float*)d_in[1];
    const float* Wk = (const float*)d_in[2];
    const float* Wv = (const float*)d_in[3];
    const float* Wo = (const float*)d_in[4];

    char* ws = (char*)d_ws;
    size_t off = 0;
    auto alloc = [&](size_t bytes) -> void* {
        void* p = ws + off;
        off += (bytes + 255) & ~(size_t)255;
        return p;
    };
    const size_t actBytes = (size_t)BT * D_MODEL * sizeof(_Float16);
    const size_t wBytes   = (size_t)D_MODEL * D_MODEL * sizeof(_Float16);
    _Float16* Xh  = (_Float16*)alloc(actBytes);
    _Float16* Wqt = (_Float16*)alloc(wBytes);   // transposed [N][K]
    _Float16* Wkt = (_Float16*)alloc(wBytes);
    _Float16* Wvt = (_Float16*)alloc(wBytes);
    _Float16* Wot = (_Float16*)alloc(wBytes);
    _Float16* Qh  = (_Float16*)alloc(actBytes);
    _Float16* Kh  = (_Float16*)alloc(actBytes);
    _Float16* Vth = (_Float16*)alloc(actBytes); // [b][h][d][t]
    _Float16* Zh  = (_Float16*)alloc(actBytes);

    const int nAct = BT * D_MODEL;
    const int nW   = D_MODEL * D_MODEL;
    cvt_f32_to_f16<<<(nAct + 255) / 256, 256, 0, stream>>>(x, Xh, nAct);
    cvt_transpose_f32_to_f16<<<(nW + 255) / 256, 256, 0, stream>>>(Wq, Wqt, D_MODEL, D_MODEL);
    cvt_transpose_f32_to_f16<<<(nW + 255) / 256, 256, 0, stream>>>(Wk, Wkt, D_MODEL, D_MODEL);
    cvt_transpose_f32_to_f16<<<(nW + 255) / 256, 256, 0, stream>>>(Wv, Wvt, D_MODEL, D_MODEL);
    cvt_transpose_f32_to_f16<<<(nW + 255) / 256, 256, 0, stream>>>(Wo, Wot, D_MODEL, D_MODEL);

    dim3 ggrid(D_MODEL / 128, BT / 128);   // (8, 64)
    gemm_f16<1><<<ggrid, 256, 0, stream>>>(Xh, Wqt, Qh,  BT, D_MODEL, D_MODEL);
    gemm_f16<1><<<ggrid, 256, 0, stream>>>(Xh, Wkt, Kh,  BT, D_MODEL, D_MODEL);
    gemm_f16<2><<<ggrid, 256, 0, stream>>>(Xh, Wvt, Vth, BT, D_MODEL, D_MODEL);

    flash_attn<<<BATCH * N_HEADS * (SEQ / 64), 128, 0, stream>>>(Qh, Kh, Vth, Zh);

    gemm_f16<0><<<ggrid, 256, 0, stream>>>(Zh, Wot, d_out, BT, D_MODEL, D_MODEL);
}
